// GDN_7868380087092
// MI455X (gfx1250) — matrix operand order repaired and checked by reading, not compiled, not run on hardware
//
#include <hip/hip_runtime.h>
#include <hip/hip_bf16.h>

typedef __attribute__((ext_vector_type(16))) _Float16 v16h;
typedef __attribute__((ext_vector_type(8)))  float    v8f;

#define N_NODES   1000
#define NPAD      1008
#define NBATCH    32
#define NBTOT     32000
#define TWIN      15
#define DDIM      64
#define KTOP      20
#define EPS_BN    1e-5f
#define INV_NB    (1.0f/32000.0f)
#define NEG_HUGE  (-3.4028234663852886e38f)

__device__ __forceinline__ float wave_sum(float v) {
  #pragma unroll
  for (int m = 16; m; m >>= 1) v += __shfl_xor(v, m, 32);
  return v;
}
__device__ __forceinline__ float wave_max(float v) {
  #pragma unroll
  for (int m = 16; m; m >>= 1) v = fmaxf(v, __shfl_xor(v, m, 32));
  return v;
}

// ---------------------------------------------------------------------------
// 0) zero the BN statistics accumulators (256 floats)
// ---------------------------------------------------------------------------
__global__ void k_zero_stats(float* stats) {
  stats[threadIdx.x] = 0.0f;
}

// ---------------------------------------------------------------------------
// 1) normalize embedding rows -> f16 (padded to NPAD rows); emb.att_em dots
// ---------------------------------------------------------------------------
__global__ void k_prep(const float* __restrict__ emb,
                       const float* __restrict__ att_em_i,
                       const float* __restrict__ att_em_j,
                       _Float16* __restrict__ wn,
                       float* __restrict__ em_i_dot,
                       float* __restrict__ em_j_dot) {
  int i = blockIdx.x;
  int L = threadIdx.x;
  if (i >= N_NODES) {
    wn[i * DDIM + L]      = (_Float16)0.0f;
    wn[i * DDIM + 32 + L] = (_Float16)0.0f;
    return;
  }
  float w0 = emb[i * DDIM + L];
  float w1 = emb[i * DDIM + 32 + L];
  float sq = wave_sum(w0 * w0 + w1 * w1);
  float inv = rsqrtf(sq);
  wn[i * DDIM + L]      = (_Float16)(w0 * inv);
  wn[i * DDIM + 32 + L] = (_Float16)(w1 * inv);
  float di = wave_sum(w0 * att_em_i[L] + w1 * att_em_i[L + 32]);
  float dj = wave_sum(w0 * att_em_j[L] + w1 * att_em_j[L + 32]);
  if (L == 0) { em_i_dot[i] = di; em_j_dot[i] = dj; }
}

// ---------------------------------------------------------------------------
// 2) cos = Wn @ Wn^T via v_wmma_f32_16x16x32_f16; grid (63,63), 32 thr/block
// ---------------------------------------------------------------------------
__global__ void k_cos_wmma(const _Float16* __restrict__ wn,
                           float* __restrict__ cosm) {
  int lane = threadIdx.x;
  int hi   = lane >> 4;
  int lr   = lane & 15;
  int m = blockIdx.x * 16 + lr;  // A row
  int n = blockIdx.y * 16 + lr;  // B col (row of Wn, since B = Wn^T)
  v8f c = {};
  #pragma unroll
  for (int kt = 0; kt < 2; ++kt) {
    v16h a, b;
    #pragma unroll
    for (int e = 0; e < 16; ++e) {
      int ka = (e < 8 ? e : e + 8) + (hi ? 8 : 0);
      a[e] = wn[m * DDIM + kt * 32 + ka];
      int kb = (e < 8 ? e : e + 8) + (hi ? 8 : 0);
      b[e] = wn[n * DDIM + kt * 32 + kb];
    }
    c = __builtin_amdgcn_wmma_f32_16x16x32_f16(false, a, false, b,
                                               (short)0, c, false, false);
  }
  #pragma unroll
  for (int r = 0; r < 8; ++r) {
    int row = blockIdx.x * 16 + r + (hi ? 8 : 0);
    cosm[row * NPAD + blockIdx.y * 16 + lr] = c[r];
  }
}

// ---------------------------------------------------------------------------
// 3) top-20 per row (descending, ties -> lower index). grid N_NODES, 256 thr
// ---------------------------------------------------------------------------
__global__ void k_topk(const float* __restrict__ cosm, int* __restrict__ topk) {
  __shared__ float vals[N_NODES];
  __shared__ unsigned long long best;
  int i = blockIdx.x, t = threadIdx.x;
  for (int j = t; j < N_NODES; j += 256) vals[j] = cosm[i * NPAD + j];
  __syncthreads();
  for (int k = 0; k < KTOP; ++k) {
    if (t == 0) best = 0ULL;
    __syncthreads();
    unsigned long long lb = 0ULL;
    for (int j = t; j < N_NODES; j += 256) {
      unsigned u = __float_as_uint(vals[j]);
      u = (u & 0x80000000u) ? ~u : (u | 0x80000000u);
      unsigned long long key = ((unsigned long long)u << 32) | (unsigned)(~j);
      if (key > lb) lb = key;
    }
    atomicMax(&best, lb);
    __syncthreads();
    if (t == 0) {
      int idx = (int)(~(unsigned)(best & 0xffffffffULL));
      topk[i * KTOP + k] = idx;
      vals[idx] = NEG_HUGE;
    }
    __syncthreads();
  }
}

// ---------------------------------------------------------------------------
// 4) xt = x @ lin_w via WMMA. grid 2000 blocks x 128 thr (4 waves = 4 N-tiles)
// ---------------------------------------------------------------------------
__global__ void k_xt_wmma(const float* __restrict__ x,       // [NBTOT, 15]
                          const float* __restrict__ lin_w,   // [15, 64]
                          float* __restrict__ xt) {          // [NBTOT, 64]
  int wave = threadIdx.x >> 5;
  int lane = threadIdx.x & 31;
  int hi = lane >> 4, lr = lane & 15;
  int m  = blockIdx.x * 16 + lr;
  int nc = wave * 16 + lr;
  v16h a, b;
  #pragma unroll
  for (int e = 0; e < 16; ++e) {
    int ka = (e < 8 ? e : e + 8) + (hi ? 8 : 0);
    a[e] = (ka < TWIN) ? (_Float16)x[m * TWIN + ka] : (_Float16)0.0f;
    int kb = (e < 8 ? e : e + 8) + (hi ? 8 : 0);
    b[e] = (kb < TWIN) ? (_Float16)lin_w[kb * DDIM + nc] : (_Float16)0.0f;
  }
  v8f c = {};
  c = __builtin_amdgcn_wmma_f32_16x16x32_f16(false, a, false, b,
                                             (short)0, c, false, false);
  #pragma unroll
  for (int r = 0; r < 8; ++r) {
    int row = blockIdx.x * 16 + r + (hi ? 8 : 0);
    xt[row * DDIM + wave * 16 + lr] = c[r];
  }
}

// ---------------------------------------------------------------------------
// 5) ai[v] = xt[v].att_i + em_i_dot[v%N]; aj likewise. 8 nodes/block
// ---------------------------------------------------------------------------
__global__ void k_aiaj(const float* __restrict__ xt,
                       const float* __restrict__ att_i,
                       const float* __restrict__ att_j,
                       const float* __restrict__ em_i_dot,
                       const float* __restrict__ em_j_dot,
                       float* __restrict__ ai, float* __restrict__ aj) {
  int wave = threadIdx.x >> 5, lane = threadIdx.x & 31;
  int v = blockIdx.x * 8 + wave;
  float x0 = xt[v * DDIM + lane], x1 = xt[v * DDIM + 32 + lane];
  float pi = wave_sum(x0 * att_i[lane] + x1 * att_i[lane + 32]);
  float pj = wave_sum(x0 * att_j[lane] + x1 * att_j[lane + 32]);
  if (lane == 0) {
    int i = v % N_NODES;
    ai[v] = pi + em_i_dot[i];
    aj[v] = pj + em_j_dot[i];
  }
}

// ---------------------------------------------------------------------------
// 6) per-node 21-way softmax attention + aggregation.
//    One block per (batch, 250-node chunk): the whole batch slice of xt
//    (1000x64 f32 = 256 KB, fits in the 320 KB WGP LDS) is staged via
//    CDNA5 async global->LDS copies; all gathers then hit LDS.
//    grid = NBATCH*4, block = 1024 (32 waves), dynamic LDS = 256000 B
// ---------------------------------------------------------------------------
__global__ void k_aggregate(const int* __restrict__ topk,
                            const float* __restrict__ ai,
                            const float* __restrict__ aj,
                            const float* __restrict__ xt,
                            const float* __restrict__ gnn_bias,
                            float* __restrict__ agg) {
  extern __shared__ float sxt[];                 // [N_NODES * DDIM]
  int b     = blockIdx.x >> 2;
  int chunk = blockIdx.x & 3;
  int t = threadIdx.x;

  // ---- stage xt[b*1000 .. +999][0..63] into LDS (async b128 copies) ----
  const float* src = xt + (size_t)b * N_NODES * DDIM;
  for (int k = t; k < (N_NODES * DDIM / 4); k += 1024) {   // 16000 x 16B
    unsigned lds_off = (unsigned)(unsigned long long)(uintptr_t)(sxt + 4 * k);
    unsigned long long ga = (unsigned long long)(uintptr_t)(src + 4 * k);
    asm volatile("global_load_async_to_lds_b128 %0, %1, off"
                 :: "v"(lds_off), "v"(ga) : "memory");
  }
  asm volatile("s_wait_asynccnt 0x0" ::: "memory");
  __syncthreads();

  int wave = t >> 5, lane = t & 31;
  for (int i = chunk * 250 + wave; i < (chunk + 1) * 250; i += 32) {
    int v = b * N_NODES + i;

    float alpha = NEG_HUGE;
    int srci = i;
    if (lane < KTOP + 1) {
      srci = (lane < KTOP) ? topk[i * KTOP + lane] : i;
      bool valid = (lane == KTOP) || (srci != i);
      float al = ai[v] + aj[b * N_NODES + srci];
      al = (al > 0.0f) ? al : 0.2f * al;          // leaky_relu(0.2)
      alpha = valid ? al : NEG_HUGE;
    }
    float mx = wave_max(alpha);
    float e  = expf(alpha - mx);                  // invalid -> 0
    float sm = wave_sum(e);
    float attn = e / (sm + 1e-16f);

    float acc0 = 0.0f, acc1 = 0.0f;
    #pragma unroll
    for (int k = 0; k < KTOP + 1; ++k) {
      float a = __shfl(attn, k, 32);
      int   s = __shfl(srci, k, 32);
      acc0 += a * sxt[s * DDIM + lane];
      acc1 += a * sxt[s * DDIM + 32 + lane];
    }
    agg[v * DDIM + lane]      = acc0 + gnn_bias[lane];
    agg[v * DDIM + 32 + lane] = acc1 + gnn_bias[lane + 32];
  }
}

// ---------------------------------------------------------------------------
// 7) BN1 stats: per-channel sum & sumsq over 32000 rows. grid 125 x 256
// ---------------------------------------------------------------------------
__global__ void k_stats1(const float* __restrict__ agg, float* __restrict__ stats) {
  __shared__ float S[4][64], Q[4][64];
  int c = threadIdx.x & 63, g = threadIdx.x >> 6;
  int base = blockIdx.x * 256 + g * 64;
  float s = 0.0f, q = 0.0f;
  for (int j = 0; j < 64; ++j) {
    float x = agg[(base + j) * DDIM + c];
    s += x; q += x * x;
  }
  S[g][c] = s; Q[g][c] = q;
  __syncthreads();
  if (g == 0) {
    float ts = S[0][c] + S[1][c] + S[2][c] + S[3][c];
    float tq = Q[0][c] + Q[1][c] + Q[2][c] + Q[3][c];
    atomicAdd(&stats[c], ts);
    atomicAdd(&stats[64 + c], tq);
  }
}

// ---------------------------------------------------------------------------
// 8) BN1 + ReLU, multiply by embedding, accumulate BN2 stats. grid 125 x 256
//    (stats read region [0..127] and write region [128..255] share a buffer:
//     no restrict qualifiers here)
// ---------------------------------------------------------------------------
__global__ void k_bn1_mul(const float* __restrict__ agg,
                          const float* __restrict__ emb,
                          const float* __restrict__ g1, const float* __restrict__ b1,
                          const float* stats,
                          float* __restrict__ out2, float* statsw) {
  __shared__ float S[4][64], Q[4][64];
  int c = threadIdx.x & 63, g = threadIdx.x >> 6;
  int base = blockIdx.x * 256 + g * 64;
  float mu  = stats[c] * INV_NB;
  float var = stats[64 + c] * INV_NB - mu * mu;
  float inv = rsqrtf(var + EPS_BN);
  float ga = g1[c], be = b1[c];
  float s = 0.0f, q = 0.0f;
  for (int j = 0; j < 64; ++j) {
    int r = base + j;
    float z = ga * (agg[r * DDIM + c] - mu) * inv + be;
    z = fmaxf(z, 0.0f);
    float o = z * emb[(r % N_NODES) * DDIM + c];
    out2[r * DDIM + c] = o;
    s += o; q += o * o;
  }
  S[g][c] = s; Q[g][c] = q;
  __syncthreads();
  if (g == 0) {
    float ts = S[0][c] + S[1][c] + S[2][c] + S[3][c];
    float tq = Q[0][c] + Q[1][c] + Q[2][c] + Q[3][c];
    atomicAdd(&statsw[128 + c], ts);
    atomicAdd(&statsw[192 + c], tq);
  }
}

// ---------------------------------------------------------------------------
// 9) BN2 + ReLU + Linear(64->1). one wave per node, 8 nodes/block
// ---------------------------------------------------------------------------
__global__ void k_final(const float* __restrict__ out2,
                        const float* __restrict__ stats,
                        const float* __restrict__ g2, const float* __restrict__ b2,
                        const float* __restrict__ out_w, const float* __restrict__ out_b,
                        float* __restrict__ y) {
  int wave = threadIdx.x >> 5, lane = threadIdx.x & 31;
  int v = blockIdx.x * 8 + wave;
  int c0 = lane, c1 = lane + 32;
  float mu0 = stats[128 + c0] * INV_NB, mu1 = stats[128 + c1] * INV_NB;
  float v0 = stats[192 + c0] * INV_NB - mu0 * mu0;
  float v1 = stats[192 + c1] * INV_NB - mu1 * mu1;
  float h0 = g2[c0] * (out2[v * DDIM + c0] - mu0) * rsqrtf(v0 + EPS_BN) + b2[c0];
  float h1 = g2[c1] * (out2[v * DDIM + c1] - mu1) * rsqrtf(v1 + EPS_BN) + b2[c1];
  h0 = fmaxf(h0, 0.0f); h1 = fmaxf(h1, 0.0f);
  float p = wave_sum(h0 * out_w[c0] + h1 * out_w[c1]);
  if (lane == 0) y[v] = p + out_b[0];
}

// ---------------------------------------------------------------------------
extern "C" void kernel_launch(void* const* d_in, const int* in_sizes, int n_in,
                              void* d_out, int out_size, void* d_ws, size_t ws_size,
                              hipStream_t stream) {
  const float* data     = (const float*)d_in[0];   // [32,1000,15] == x [32000,15]
  const float* emb      = (const float*)d_in[1];   // [1000,64]
  const float* lin_w    = (const float*)d_in[2];   // [15,64]
  const float* att_i    = (const float*)d_in[3];
  const float* att_j    = (const float*)d_in[4];
  const float* att_em_i = (const float*)d_in[5];
  const float* att_em_j = (const float*)d_in[6];
  const float* gnn_bias = (const float*)d_in[7];
  const float* bn1_g    = (const float*)d_in[8];
  const float* bn1_b    = (const float*)d_in[9];
  const float* bn2_g    = (const float*)d_in[10];
  const float* bn2_b    = (const float*)d_in[11];
  const float* out_w    = (const float*)d_in[12];
  const float* out_b    = (const float*)d_in[13];
  float* y = (float*)d_out;

  char* ws = (char*)d_ws;
  size_t off = 0;
  auto carve = [&](size_t bytes) -> char* {
    char* p = ws + off;
    off = (off + bytes + 255) & ~(size_t)255;
    return p;
  };
  _Float16* wn     = (_Float16*)carve((size_t)NPAD * DDIM * 2);
  float* em_i_dot  = (float*)carve((size_t)N_NODES * 4);
  float* em_j_dot  = (float*)carve((size_t)N_NODES * 4);
  float* cosm      = (float*)carve((size_t)NPAD * NPAD * 4);
  int*   topk      = (int*)  carve((size_t)N_NODES * KTOP * 4);
  float* xt        = (float*)carve((size_t)NBTOT * DDIM * 4);
  float* ai        = (float*)carve((size_t)NBTOT * 4);
  float* aj        = (float*)carve((size_t)NBTOT * 4);
  float* agg       = (float*)carve((size_t)NBTOT * DDIM * 4);
  float* out2      = (float*)carve((size_t)NBTOT * DDIM * 4);
  float* stats     = (float*)carve(256 * 4);

  k_zero_stats<<<1, 256, 0, stream>>>(stats);
  k_prep<<<NPAD, 32, 0, stream>>>(emb, att_em_i, att_em_j, wn, em_i_dot, em_j_dot);
  k_cos_wmma<<<dim3(NPAD / 16, NPAD / 16), 32, 0, stream>>>(wn, cosm);
  k_topk<<<N_NODES, 256, 0, stream>>>(cosm, topk);
  k_xt_wmma<<<NBTOT / 16, 128, 0, stream>>>(data, lin_w, xt);
  k_aiaj<<<NBTOT / 8, 256, 0, stream>>>(xt, att_i, att_j, em_i_dot, em_j_dot, ai, aj);
  k_aggregate<<<NBATCH * 4, 1024, (size_t)N_NODES * DDIM * 4, stream>>>(
      topk, ai, aj, xt, gnn_bias, agg);
  k_stats1<<<NBTOT / 256, 256, 0, stream>>>(agg, stats);
  k_bn1_mul<<<NBTOT / 256, 256, 0, stream>>>(agg, emb, bn1_g, bn1_b, stats, out2, stats);
  k_final<<<NBTOT / 8, 256, 0, stream>>>(out2, stats, bn2_g, bn2_b, out_w, out_b, y);
}
